// MalGAT_37580963840176
// MI455X (gfx1250) — matrix operand order repaired and checked by reading, not compiled, not run on hardware
//
#include <hip/hip_runtime.h>

#define K_ 2
#define B_ 8
#define V_ 1024
#define D_ 64
#define HID_ 64
#define H_ 2
#define P_ 64
#define F_ 128
#define NEG_ -9e15f
#define ALPHA_ 0.2f

#define BROW_ 1032   // padded LDS row stride (halves) for 1024-wide B rows
#define WROW_ 136    // padded LDS row stride (halves) for 128-wide out_w^T rows

typedef __attribute__((ext_vector_type(16))) _Float16 v16h;
typedef __attribute__((ext_vector_type(8)))  _Float16 v8h;
typedef __attribute__((ext_vector_type(8)))  float    v8f;
typedef __attribute__((ext_vector_type(4)))  int      v4i;

#if defined(__has_builtin)
#if __has_builtin(__builtin_amdgcn_global_load_async_to_lds_b128)
#define HAVE_ASYNC_LDS 1
#endif
#if __has_builtin(__builtin_amdgcn_s_wait_asynccnt)
#define HAVE_WAIT_ASYNC_BUILTIN 1
#endif
#endif
#ifndef HAVE_ASYNC_LDS
#define HAVE_ASYNC_LDS 0
#endif
#ifndef HAVE_WAIT_ASYNC_BUILTIN
#define HAVE_WAIT_ASYNC_BUILTIN 0
#endif

__device__ __forceinline__ float lreluf(float z) { return z > 0.f ? z : ALPHA_ * z; }
__device__ __forceinline__ float eluf(float z)   { return z > 0.f ? z : (__expf(z) - 1.f); }

__device__ __forceinline__ v8f wmma16(v16h a, v16h b, v8f c) {
  // D = A(16x32 f16) * B(32x16 f16) + C(16x16 f32)
  return __builtin_amdgcn_wmma_f32_16x16x32_f16(false, a, false, b, (short)0, c,
                                                false, false);
}

// 16-byte global -> LDS copy; async (ASYNCcnt) when the gfx1250 builtin exists.
// Builtin signature (from probe): (v4i AS1* global, v4i AS3* lds, imm off, imm cpol)
__device__ __forceinline__ void copy_b128_to_lds(const _Float16* g, _Float16* l) {
#if HAVE_ASYNC_LDS
  __builtin_amdgcn_global_load_async_to_lds_b128(
      (__attribute__((address_space(1))) v4i*)(__attribute__((address_space(1))) void*)g,
      (__attribute__((address_space(3))) v4i*)(__attribute__((address_space(3))) void*)l,
      0, 0);
#else
  *(v8h*)l = *(const v8h*)g;
#endif
}

__device__ __forceinline__ void wait_async_lds() {
#if HAVE_ASYNC_LDS
#if HAVE_WAIT_ASYNC_BUILTIN
  __builtin_amdgcn_s_wait_asynccnt(0);
#else
  asm volatile("s_wait_asynccnt 0x0" ::: "memory");
#endif
#endif
}

__device__ __forceinline__ v16h lds_load_v16h(const _Float16* l) {
  v8h lo = *(const v8h*)l;
  v8h hi = *(const v8h*)(l + 8);
  v16h r;
#pragma unroll
  for (int i = 0; i < 8; ++i) { r[i] = lo[i]; r[8 + i] = hi[i]; }
  return r;
}

// ---------------------------------------------------------------------------
// Tiny prep kernels
// ---------------------------------------------------------------------------
__global__ void prep_xc_kernel(const float* __restrict__ x, float* __restrict__ XC) {
  int i = blockIdx.x * 256 + threadIdx.x;              // B*V = 8192
  XC[i] = fminf(x[i] + x[B_ * V_ + i], 1.f);           // clip(sum_k x, 0, 1), x >= 0
}

__global__ void prep_awt_kernel(const float* __restrict__ attn_w, _Float16* __restrict__ AWT) {
  int i = blockIdx.x * 256 + threadIdx.x;              // 64*1024, i = d*V + v
  int d = i >> 10, v = i & (V_ - 1);
  AWT[i] = (_Float16)attn_w[v * 64 + d];               // transpose [V,D] -> [D][V]
}

__global__ void prep_owt_kernel(const float* __restrict__ out_w, _Float16* __restrict__ OWT) {
  int i = blockIdx.x * 256 + threadIdx.x;              // 64*128, i = p*F + f
  int pp = i >> 7, f = i & (F_ - 1);
  OWT[i] = (_Float16)out_w[f * 64 + pp];               // transpose [F,P] -> [P][F]
}

// EW[h] = emb @ gat_w[h]; store transposed f16 [H][HID][V] + score dots es0/ed0.
__global__ __launch_bounds__(256)
void prep_ew_kernel(const float* __restrict__ emb, const float* __restrict__ gat_w,
                    const float* __restrict__ gat_a,
                    _Float16* __restrict__ EWT, float* __restrict__ ES0,
                    float* __restrict__ ED0) {
  int h = blockIdx.x >> 8, vblk = blockIdx.x & 255;
  int e = threadIdx.x, vl = threadIdx.y;
  int v = vblk * 4 + vl;
  __shared__ float sw[64 * 64];
  __shared__ float se[4][64];
  __shared__ float sa[4][64];
  int tid = vl * 64 + e;
#pragma unroll
  for (int i = 0; i < 16; ++i) sw[tid + i * 256] = gat_w[h * 4096 + tid + i * 256];
  se[vl][e] = emb[v * 64 + e];
  __syncthreads();
  float acc = 0.f;
#pragma unroll 8
  for (int d = 0; d < 64; ++d) acc += se[vl][d] * sw[d * 64 + e];
  EWT[(size_t)h * 64 * V_ + (size_t)e * V_ + v] = (_Float16)acc;
  sa[vl][e] = acc;
  __syncthreads();
  if (e == 0) {
    float s0 = 0.f, s1 = 0.f;
    for (int ee = 0; ee < 64; ++ee) {
      float a = sa[vl][ee];
      s0 += a * gat_a[h * 128 + ee];
      s1 += a * gat_a[h * 128 + 64 + ee];
    }
    ES0[h * V_ + v] = s0;
    ED0[h * V_ + v] = s1;
  }
}

// ---------------------------------------------------------------------------
// Modality-frequency branch: m[b,d,e] = sum_v xc*emb[v,d]*frq_w[v,e]
// ---------------------------------------------------------------------------
__global__ __launch_bounds__(256)
void freq_kernel(const float* __restrict__ XC, const float* __restrict__ emb,
                 const float* __restrict__ frq_w, float* __restrict__ MW) {
  int b = blockIdx.x;
  int e = blockIdx.y * 4 + threadIdx.y;
  int d = threadIdx.x;
  __shared__ float xc[V_];
  int tid = threadIdx.y * 64 + threadIdx.x;
#pragma unroll
  for (int i = 0; i < 4; ++i) xc[tid + i * 256] = XC[b * V_ + tid + i * 256];
  __syncthreads();
  float acc = 0.f;
  for (int v = 0; v < V_; ++v) {
    float xv = xc[v];
    if (xv != 0.f) acc += xv * emb[v * 64 + d] * frq_w[v * 64 + e];
  }
  MW[b * 4096 + d * 64 + e] = acc;
}

// mod1 = elu(max_e(m + frq_b)); cls_dense = mod1@cls_w + cls_b; cls_code = elu(.)
__global__ __launch_bounds__(64)
void mod1cls_kernel(const float* __restrict__ MW, const float* __restrict__ frq_b,
                    const float* __restrict__ cls_w, const float* __restrict__ cls_b,
                    float* __restrict__ CLSD, float* __restrict__ CLSC) {
  int b = blockIdx.x, t = threadIdx.x;
  float mx = -3.4e38f;
  for (int e = 0; e < 64; ++e) mx = fmaxf(mx, MW[b * 4096 + t * 64 + e] + frq_b[e]);
  __shared__ float md[64];
  md[t] = eluf(mx);                                    // elu monotonic -> elu(max)
  __syncthreads();
  float cd = cls_b[t];
  for (int d = 0; d < 64; ++d) cd += md[d] * cls_w[d * 64 + t];
  CLSD[b * 64 + t] = cd;
  CLSC[b * 64 + t] = eluf(cd);
}

// ---------------------------------------------------------------------------
// Fused masked-softmax attention + WMMA GEMM.
// e[u,v] = mask ? lrelu(s_u + t_v) : NEG ; A[u,v] = x_v * exp(e-m_u)/Z_u
// out = A @ Bt^T where Bt is [64][V] f16 (EW^T for layer1, Who^T for layer2).
// Bt is staged to LDS via async global->LDS copies overlapped with phase 1.
// ---------------------------------------------------------------------------
template <bool LAYER2>
__global__ __launch_bounds__(256)
void attn_kernel(const float* __restrict__ x, const float* __restrict__ sArr,
                 const float* __restrict__ tArr, const _Float16* __restrict__ BtArr,
                 _Float16* __restrict__ out) {
  int bid = blockIdx.y;
  int kb = LAYER2 ? bid : (bid >> 1);
  int h  = LAYER2 ? 0 : (bid & 1);
  int unit = LAYER2 ? kb : h;
  const float* s = sArr + unit * V_;
  const float* t = tArr + unit * V_;
  const _Float16* Bt = BtArr + (size_t)unit * 64 * V_;
  const float* xr = x + kb * V_;

  __shared__ _Float16 shB[64 * BROW_];   // 132 KB, padded rows (bank-conflict free)
  __shared__ float sh_x[V_];
  __shared__ float sh_t[V_];
  __shared__ float sh_m[8][16];
  __shared__ float sh_zr[8][16];

  int tid = threadIdx.x;

  // ---- kick off B staging (ASYNCcnt path), overlapped with phase 1 ----
  for (int c = tid; c < 64 * 128; c += 256) {          // 8192 x 16B chunks
    int row = c >> 7, coloff = (c & 127) * 8;          // offsets in halves
    copy_b128_to_lds(Bt + (size_t)row * V_ + coloff, &shB[row * BROW_ + coloff]);
  }
  for (int i = tid; i < V_; i += 256) { sh_x[i] = xr[i]; sh_t[i] = t[i]; }
  __syncthreads();                                     // x/t ready (phase 1 input)

  int w = tid >> 5, lane = tid & 31;
  int u0 = blockIdx.x * 128 + w * 16;

  // ---- phase 1: per-row softmax stats (two-pass, lanes stride columns) ----
  float su[16], xu[16], mr[16];
#pragma unroll
  for (int r = 0; r < 16; ++r) { su[r] = s[u0 + r]; xu[r] = sh_x[u0 + r]; mr[r] = NEG_; }
  for (int v = lane; v < V_; v += 32) {
    float xv = sh_x[v], tv = sh_t[v];
    bool av = xv > 0.f;
#pragma unroll
    for (int r = 0; r < 16; ++r) {
      float e = (av && xu[r] > 0.f) ? lreluf(su[r] + tv) : NEG_;
      mr[r] = fmaxf(mr[r], e);
    }
  }
#pragma unroll
  for (int r = 0; r < 16; ++r) {
#pragma unroll
    for (int off = 16; off >= 1; off >>= 1) mr[r] = fmaxf(mr[r], __shfl_xor(mr[r], off, 32));
  }
  float zr[16];
#pragma unroll
  for (int r = 0; r < 16; ++r) zr[r] = 0.f;
  for (int v = lane; v < V_; v += 32) {
    float xv = sh_x[v], tv = sh_t[v];
    bool av = xv > 0.f;
#pragma unroll
    for (int r = 0; r < 16; ++r) {
      float e = (av && xu[r] > 0.f) ? lreluf(su[r] + tv) : NEG_;
      zr[r] += __expf(e - mr[r]);                      // all-masked row: exp(0)=1 each
    }
  }
#pragma unroll
  for (int r = 0; r < 16; ++r) {
#pragma unroll
    for (int off = 16; off >= 1; off >>= 1) zr[r] += __shfl_xor(zr[r], off, 32);
  }
  if (lane == 0) {
#pragma unroll
    for (int r = 0; r < 16; ++r) { sh_m[w][r] = mr[r]; sh_zr[w][r] = 1.f / zr[r]; }
  }
  wait_async_lds();                                    // own async copies landed
  __syncthreads();                                     // all waves' copies visible

  // ---- phase 2: GEMM, A regenerated on the fly per 16x32 K-chunk ----
  int col = lane & 15, khalf = lane >> 4;
  float msu = s[u0 + col];
  float mxu = sh_x[u0 + col];
  float mmr = sh_m[w][col];
  float mzr = sh_zr[w][col];
  v8f czero = {};
  v8f c[4] = {czero, czero, czero, czero};
  for (int kc = 0; kc < 32; ++kc) {
    int vb = kc * 32;
    v16h a;
#pragma unroll
    for (int e = 0; e < 16; ++e) {
      // 16-bit A 16x32 layout: K = (e<8?0:16) + khalf*8 + (e&7)
      int v = vb + ((e < 8) ? 0 : 16) + khalf * 8 + (e & 7);
      float xv = sh_x[v], tv = sh_t[v];
      float ev = (mxu > 0.f && xv > 0.f) ? lreluf(msu + tv) : NEG_;
      float pval = xv * __expf(ev - mmr) * mzr;        // x folded into A
      a[e] = (_Float16)pval;
    }
#pragma unroll
    for (int n = 0; n < 4; ++n) {
      // B 32x16 layout: K = (e<8?0:16) + khalf*8 + (e&7), matching A
      const _Float16* bp = &shB[(n * 16 + col) * BROW_ + vb + khalf * 8];
      v8h blo = *(const v8h*)bp;
      v8h bhi = *(const v8h*)(bp + 16);
      v16h b;
#pragma unroll
      for (int i = 0; i < 8; ++i) { b[i] = blo[i]; b[8 + i] = bhi[i]; }
      c[n] = wmma16(a, b, c[n]);
    }
  }

  // ---- epilogue ----
  int ub = u0 + khalf * 8;                             // 8 consecutive rows per lane
  if (!LAYER2) {
    // feats1 [KB][V][F] row-major f16, columns h*64 + n*16 + col, elu applied
#pragma unroll
    for (int n = 0; n < 4; ++n) {
      int cg = h * 64 + n * 16 + col;
#pragma unroll
      for (int i = 0; i < 8; ++i)
        out[((size_t)(kb * V_) + (ub + i)) * F_ + cg] = (_Float16)eluf(c[n][i]);
    }
  } else {
    // feats2^T [KB][P][V] f16, value = x_u * elu(.), packed b128 stores
#pragma unroll
    for (int n = 0; n < 4; ++n) {
      int pcol = n * 16 + col;
      v8h st;
#pragma unroll
      for (int i = 0; i < 8; ++i) st[i] = (_Float16)(sh_x[ub + i] * eluf(c[n][i]));
      *(v8h*)(out + ((size_t)(kb * 64) + pcol) * V_ + ub) = st;
    }
  }
}

// ---------------------------------------------------------------------------
// Who = feats1 @ out_w ([V,128]@[128,64]) per (k,b); store Who^T f16 and the
// two per-row score dots with out_a. out_w^T staged to LDS asynchronously
// while the A fragments load into registers.
// ---------------------------------------------------------------------------
__global__ __launch_bounds__(256)
void who_kernel(const _Float16* __restrict__ F1, const _Float16* __restrict__ OWT,
                const float* __restrict__ out_a, _Float16* __restrict__ WHOT,
                float* __restrict__ SSRC, float* __restrict__ SDST) {
  int kb = blockIdx.y;
  int tid = threadIdx.x, w = tid >> 5, lane = tid & 31;
  int col = lane & 15, khalf = lane >> 4;
  int u0 = blockIdx.x * 128 + w * 16;

  __shared__ _Float16 shW[64 * WROW_];                 // 17 KB padded out_w^T
  for (int c = tid; c < 64 * 16; c += 256) {           // 1024 x 16B chunks
    int row = c >> 4, coloff = (c & 15) * 8;
    copy_b128_to_lds(OWT + (size_t)row * F_ + coloff, &shW[row * WROW_ + coloff]);
  }

  // A fragments (full K=128) into registers, overlapping the async staging
  const _Float16* Ar = F1 + ((size_t)kb * V_ + u0 + col) * F_;
  v16h a[4];
#pragma unroll
  for (int kc = 0; kc < 4; ++kc) {
    v8h alo = *(const v8h*)(Ar + kc * 32 + khalf * 8);
    v8h ahi = *(const v8h*)(Ar + kc * 32 + 16 + khalf * 8);
#pragma unroll
    for (int i = 0; i < 8; ++i) { a[kc][i] = alo[i]; a[kc][8 + i] = ahi[i]; }
  }
  wait_async_lds();
  __syncthreads();

  v8f czero = {};
  v8f c[4] = {czero, czero, czero, czero};
#pragma unroll
  for (int kc = 0; kc < 4; ++kc) {
#pragma unroll
    for (int n = 0; n < 4; ++n) {
      const _Float16* bp = &shW[(n * 16 + col) * WROW_ + kc * 32 + khalf * 8];
      v8h blo = *(const v8h*)bp;
      v8h bhi = *(const v8h*)(bp + 16);
      v16h b;
#pragma unroll
      for (int i = 0; i < 8; ++i) { b[i] = blo[i]; b[8 + i] = bhi[i]; }
      c[n] = wmma16(a[kc], b, c[n]);
    }
  }

  int ub = u0 + khalf * 8;
  float ps[8], pd[8];
#pragma unroll
  for (int i = 0; i < 8; ++i) { ps[i] = 0.f; pd[i] = 0.f; }
#pragma unroll
  for (int n = 0; n < 4; ++n) {
    int cg = n * 16 + col;
    float as = out_a[cg], ad = out_a[64 + cg];
    v8h st;
#pragma unroll
    for (int i = 0; i < 8; ++i) {
      float wv = c[n][i];
      st[i] = (_Float16)wv;
      ps[i] += wv * as;
      pd[i] += wv * ad;
    }
    *(v8h*)(WHOT + ((size_t)(kb * 64) + cg) * V_ + ub) = st;
  }
#pragma unroll
  for (int off = 8; off >= 1; off >>= 1) {
#pragma unroll
    for (int i = 0; i < 8; ++i) {
      ps[i] += __shfl_xor(ps[i], off, 16);
      pd[i] += __shfl_xor(pd[i], off, 16);
    }
  }
  if (col == 0) {
#pragma unroll
    for (int i = 0; i < 8; ++i) {
      SSRC[kb * V_ + ub + i] = ps[i];
      SDST[kb * V_ + ub + i] = pd[i];
    }
  }
}

// ---------------------------------------------------------------------------
// Pooling: acde[p,d] = sum_v (x*feats2)[v,p]*attn_w[v,d]; code = elu(max_d(.+b))
// ---------------------------------------------------------------------------
__global__ __launch_bounds__(128)
void pool_kernel(const _Float16* __restrict__ F2T, const _Float16* __restrict__ AWT,
                 const float* __restrict__ attn_b, float* __restrict__ ACODE) {
  int kb = blockIdx.y, pt = blockIdx.x;
  int tid = threadIdx.x, w = tid >> 5, lane = tid & 31;  // wave = d-tile
  int col = lane & 15, khalf = lane >> 4;
  const _Float16* Ar = F2T + ((size_t)(kb * 64) + pt * 16 + col) * V_;
  const _Float16* Br = AWT + (size_t)(w * 16 + col) * V_;
  v8f c = {};
  for (int kc = 0; kc < 32; ++kc) {
    v8h alo = *(const v8h*)(Ar + kc * 32 + khalf * 8);
    v8h ahi = *(const v8h*)(Ar + kc * 32 + 16 + khalf * 8);
    v16h a;
#pragma unroll
    for (int i = 0; i < 8; ++i) { a[i] = alo[i]; a[8 + i] = ahi[i]; }
    v8h blo = *(const v8h*)(Br + kc * 32 + khalf * 8);
    v8h bhi = *(const v8h*)(Br + kc * 32 + 16 + khalf * 8);
    v16h b;
#pragma unroll
    for (int i = 0; i < 8; ++i) { b[i] = blo[i]; b[8 + i] = bhi[i]; }
    c = wmma16(a, b, c);
  }
  float ab = attn_b[w * 16 + col];
  float y[8];
#pragma unroll
  for (int i = 0; i < 8; ++i) y[i] = c[i] + ab;
#pragma unroll
  for (int off = 8; off >= 1; off >>= 1) {
#pragma unroll
    for (int i = 0; i < 8; ++i) y[i] = fmaxf(y[i], __shfl_xor(y[i], off, 16));
  }
  __shared__ float red[4][16];
  if (col == 0) {
#pragma unroll
    for (int i = 0; i < 8; ++i) red[w][khalf * 8 + i] = y[i];
  }
  __syncthreads();
  if (tid < 16) {
    float mx = fmaxf(fmaxf(red[0][tid], red[1][tid]), fmaxf(red[2][tid], red[3][tid]));
    ACODE[kb * 64 + pt * 16 + tid] = eluf(mx);
  }
}

// ---------------------------------------------------------------------------
// CLS attention over the k graphs + fusion (tiny).
// ---------------------------------------------------------------------------
__global__ __launch_bounds__(64)
void final_kernel(const float* __restrict__ ACODE, const float* __restrict__ CLSD,
                  const float* __restrict__ CLSC, const float* __restrict__ clsatt_w,
                  const float* __restrict__ clsatt_a, float* __restrict__ outp) {
  int b = blockIdx.x, q = threadIdx.x;
  __shared__ float lat[2][64], cc[64], whl[2][2][64], wcl[2][64];
  __shared__ float red[64];
  __shared__ float ec[2][2];
  lat[0][q] = ACODE[(0 * B_ + b) * 64 + q];
  lat[1][q] = ACODE[(1 * B_ + b) * 64 + q];
  cc[q] = CLSC[b * 64 + q];
  __syncthreads();
#pragma unroll
  for (int h = 0; h < 2; ++h) {
    float wc = 0.f, w0 = 0.f, w1 = 0.f;
    for (int p = 0; p < 64; ++p) {
      float wv = clsatt_w[h * 4096 + p * 64 + q];
      wc += cc[p] * wv;
      w0 += lat[0][p] * wv;
      w1 += lat[1][p] * wv;
    }
    wcl[h][q] = wc;
    whl[h][0][q] = w0;
    whl[h][1][q] = w1;
  }
  __syncthreads();
#pragma unroll
  for (int h = 0; h < 2; ++h) {
    red[q] = wcl[h][q] * clsatt_a[h * 128 + q];
    __syncthreads();
    if (q == 0) { float s = 0.f; for (int i = 0; i < 64; ++i) s += red[i]; red[0] = s; }
    __syncthreads();
    float ssrc = red[0];
    __syncthreads();
    red[q] = whl[h][0][q] * clsatt_a[h * 128 + 64 + q];
    __syncthreads();
    if (q == 0) { float s = 0.f; for (int i = 0; i < 64; ++i) s += red[i]; ec[h][0] = lreluf(ssrc + s); }
    __syncthreads();
    red[q] = whl[h][1][q] * clsatt_a[h * 128 + 64 + q];
    __syncthreads();
    if (q == 0) { float s = 0.f; for (int i = 0; i < 64; ++i) s += red[i]; ec[h][1] = lreluf(ssrc + s); }
    __syncthreads();
  }
  float outc = 0.f;
#pragma unroll
  for (int h = 0; h < 2; ++h) {
    float e0 = ec[h][0], e1 = ec[h][1];
    float mm = fmaxf(e0, e1);
    float x0 = __expf(e0 - mm), x1 = __expf(e1 - mm);
    float inv = 1.f / (x0 + x1);
    outc += x0 * inv * whl[h][0][q] + x1 * inv * whl[h][1][q];
  }
  outc *= 0.5f;  // / Hc
  outp[b * 64 + q] = eluf(outc + CLSD[b * 64 + q]);
}

// ---------------------------------------------------------------------------
extern "C" void kernel_launch(void* const* d_in, const int* in_sizes, int n_in,
                              void* d_out, int out_size, void* d_ws, size_t ws_size,
                              hipStream_t stream) {
  (void)in_sizes; (void)n_in; (void)out_size; (void)ws_size;
  const float* x        = (const float*)d_in[0];
  const float* emb      = (const float*)d_in[1];
  const float* gat_w    = (const float*)d_in[2];
  const float* gat_a    = (const float*)d_in[3];
  const float* out_w    = (const float*)d_in[4];
  const float* out_a    = (const float*)d_in[5];
  const float* attn_w   = (const float*)d_in[6];
  const float* attn_b   = (const float*)d_in[7];
  const float* frq_w    = (const float*)d_in[8];
  const float* frq_b    = (const float*)d_in[9];
  const float* cls_w    = (const float*)d_in[10];
  const float* cls_b    = (const float*)d_in[11];
  const float* clsatt_w = (const float*)d_in[12];
  const float* clsatt_a = (const float*)d_in[13];
  float* outp = (float*)d_out;

  char* p = (char*)d_ws;
  auto alloc = [&](size_t bytes) {
    char* r = p;
    p += (bytes + 255) & ~(size_t)255;
    return r;
  };
  _Float16* EWT  = (_Float16*)alloc((size_t)H_ * 64 * V_ * 2);        // [H][HID][V]
  float*    ES0  = (float*)alloc((size_t)H_ * V_ * 4);
  float*    ED0  = (float*)alloc((size_t)H_ * V_ * 4);
  float*    XC   = (float*)alloc((size_t)B_ * V_ * 4);
  _Float16* AWT  = (_Float16*)alloc((size_t)64 * V_ * 2);             // attn_w^T
  _Float16* OWT  = (_Float16*)alloc((size_t)64 * F_ * 2);             // out_w^T
  _Float16* F1   = (_Float16*)alloc((size_t)K_ * B_ * V_ * F_ * 2);   // feats1
  _Float16* WHOT = (_Float16*)alloc((size_t)K_ * B_ * 64 * V_ * 2);   // Who^T
  float*    SSRC = (float*)alloc((size_t)K_ * B_ * V_ * 4);
  float*    SDST = (float*)alloc((size_t)K_ * B_ * V_ * 4);
  _Float16* F2T  = (_Float16*)alloc((size_t)K_ * B_ * 64 * V_ * 2);   // (x*feats2)^T
  float*    MW   = (float*)alloc((size_t)B_ * 64 * 64 * 4);
  float*    CLSD = (float*)alloc((size_t)B_ * 64 * 4);
  float*    CLSC = (float*)alloc((size_t)B_ * 64 * 4);
  float*    ACODE= (float*)alloc((size_t)K_ * B_ * 64 * 4);

  prep_xc_kernel<<<32, 256, 0, stream>>>(x, XC);
  prep_awt_kernel<<<256, 256, 0, stream>>>(attn_w, AWT);
  prep_owt_kernel<<<32, 256, 0, stream>>>(out_w, OWT);
  prep_ew_kernel<<<512, dim3(64, 4), 0, stream>>>(emb, gat_w, gat_a, EWT, ES0, ED0);
  freq_kernel<<<dim3(B_, 16), dim3(64, 4), 0, stream>>>(XC, emb, frq_w, MW);
  mod1cls_kernel<<<B_, 64, 0, stream>>>(MW, frq_b, cls_w, cls_b, CLSD, CLSC);
  attn_kernel<false><<<dim3(8, K_ * B_ * H_), 256, 0, stream>>>(x, ES0, ED0, EWT, F1);
  who_kernel<<<dim3(8, K_ * B_), 256, 0, stream>>>(F1, OWT, out_a, WHOT, SSRC, SDST);
  attn_kernel<true><<<dim3(8, K_ * B_), 256, 0, stream>>>(x, SSRC, SDST, WHOT, F2T);
  pool_kernel<<<dim3(4, K_ * B_), 128, 0, stream>>>(F2T, AWT, attn_b, ACODE);
  final_kernel<<<B_, 64, 0, stream>>>(ACODE, CLSD, CLSC, clsatt_w, clsatt_a, outp);
}
